// dual_channel_attention_branches_16767552323861
// MI455X (gfx1250) — compile-verified, matrix operand order checked by reading, not run on hardware
//
#include <hip/hip_runtime.h>
#include <hip/hip_bf16.h>
#include <math.h>

// ---------------------------------------------------------------------------
// Dual-branch transformer block for MI455X (gfx1250, wave32, WMMA).
// GEMMs on v_wmma_f32_16x16x32_f16 (f32 accumulate); operand tiles staged in
// LDS with GLOBAL_LOAD_ASYNC_TO_LDS_B128 (+ s_wait_asynccnt) when available.
// ---------------------------------------------------------------------------

typedef _Float16 half_t;
typedef __attribute__((ext_vector_type(16))) _Float16 v16h;
typedef __attribute__((ext_vector_type(8)))  float    v8f;
typedef __attribute__((ext_vector_type(4)))  int      v4i;

#define B_   8
#define N_   1024
#define C_   768
#define H_   12
#define D_   64
#define HID3 3072
#define M_   (B_ * N_)   // 8192 rows

// ---- async copy (global -> LDS), 16 bytes per lane ------------------------
// Builtin signature (from clang diagnostic): first param is int4* in the
// global (__device__/addrspace(1)) AS; LDS dest is addrspace(3).
typedef __attribute__((address_space(1))) v4i gv4i_t;
typedef __attribute__((address_space(3))) v4i lv4i_t;

#if defined(__HIP_DEVICE_COMPILE__) && __has_builtin(__builtin_amdgcn_global_load_async_to_lds_b128)
#define ASYNC_CP16(ldsp, gp)                                                   \
  __builtin_amdgcn_global_load_async_to_lds_b128((gv4i_t*)(gp),                \
                                                 (lv4i_t*)(ldsp), 0, 0)
#if __has_builtin(__builtin_amdgcn_s_wait_asynccnt)
#define ASYNC_WAIT() __builtin_amdgcn_s_wait_asynccnt(0)
#else
#define ASYNC_WAIT() asm volatile("s_wait_asynccnt 0x0" ::: "memory")
#endif
#else
#define ASYNC_CP16(ldsp, gp)                                                   \
  do { *reinterpret_cast<uint4*>(ldsp) = *reinterpret_cast<const uint4*>(gp); } while (0)
#define ASYNC_WAIT() do { } while (0)
#endif

// ---- fragment loader ------------------------------------------------------
// A-matrix 16x32 f16 layout (ISA 7.12.2): lane holds row = lane%16;
// element j = 2v+p maps to K = 2v+p + (v>=4 ? 8 : 0) + (lane>=16 ? 8 : 0).
// K-pairs are contiguous -> 8 dword loads per fragment (global or DS).
__device__ __forceinline__ v16h ld_frag_row(const half_t* base, int k0, int lane) {
  union { v16h v; uint32_t u[8]; } f;
  const int hi = (lane & 16) ? 8 : 0;
#pragma unroll
  for (int i = 0; i < 8; ++i) {
    int k = k0 + 2 * i + ((i >= 4) ? 8 : 0) + hi;
    f.u[i] = *reinterpret_cast<const uint32_t*>(base + k);
  }
  return f.v;
}

// ---- fp32 -> fp16 conversion ----------------------------------------------
__global__ void f32_to_f16_kernel(const float* __restrict__ in,
                                  half_t* __restrict__ out, int n) {
  int i = blockIdx.x * blockDim.x + threadIdx.x;
  if (i < n) out[i] = (half_t)in[i];
}

// ---- V transpose: qkv [B*N,3C] -> vt [B*H, D, N] (f16) --------------------
__global__ __launch_bounds__(256)
void v_transpose_kernel(const half_t* __restrict__ qkv, half_t* __restrict__ vt) {
  const int idx = blockIdx.x * 256 + threadIdx.x;   // over B*H*D*N
  const int n = idx & (N_ - 1);
  const int r = idx >> 10;          // bh*D + d
  const int d = r & (D_ - 1);
  const int bh = r >> 6;
  const int b = bh / H_;
  const int h = bh % H_;
  vt[idx] = qkv[(size_t)(b * N_ + n) * (3 * C_) + 2 * C_ + h * D_ + d];
}

// ---- LayerNorm: one row (C=768) per 256-thread block, f16 output ----------
__global__ __launch_bounds__(256)
void layernorm_f16_kernel(const float* __restrict__ x, const float* __restrict__ g,
                          const float* __restrict__ bta, half_t* __restrict__ out) {
  __shared__ float s1[256], s2[256];
  const int row = blockIdx.x;
  const int t = threadIdx.x;
  const float* xr = x + (size_t)row * C_;
  float a0 = xr[t], a1 = xr[t + 256], a2 = xr[t + 512];
  s1[t] = a0 + a1 + a2;
  s2[t] = a0 * a0 + a1 * a1 + a2 * a2;
  __syncthreads();
  for (int s = 128; s > 0; s >>= 1) {
    if (t < s) { s1[t] += s1[t + s]; s2[t] += s2[t + s]; }
    __syncthreads();
  }
  const float mu = s1[0] * (1.0f / C_);
  const float var = s2[0] * (1.0f / C_) - mu * mu;
  const float rs = rsqrtf(var + 1e-5f);
  half_t* o = out + (size_t)row * C_;
  o[t]       = (half_t)((a0 - mu) * rs * g[t]       + bta[t]);
  o[t + 256] = (half_t)((a1 - mu) * rs * g[t + 256] + bta[t + 256]);
  o[t + 512] = (half_t)((a2 - mu) * rs * g[t + 512] + bta[t + 512]);
}

// ---- WMMA GEMM: out[M,Nout] = A[M,K] * W[Nout,K]^T (+bias)(+gelu)(+resid)
// Block (8 waves) computes a 128x64 tile; the shared 64x32 W strip is staged
// through LDS per k-step via async copies; B-fragments come from LDS.
template <bool OUT_F16, bool GELU, bool RESID>
__global__ __launch_bounds__(256)
void gemm_wmma_kernel(const half_t* __restrict__ A, const half_t* __restrict__ W,
                      const float* __restrict__ bias, const float* __restrict__ resid,
                      float* __restrict__ outF, half_t* __restrict__ outH,
                      int Mrows, int Nout, int K) {
  __shared__ __align__(16) half_t wtile[64 * 32];
  const int tid = threadIdx.x;
  const int lane = tid & 31;
  const int wave = tid >> 5;
  const int ntn = Nout >> 6;
  const int m0 = (blockIdx.x / ntn) * 128 + wave * 16;
  const int n0 = (blockIdx.x % ntn) * 64;
  (void)Mrows;

  const half_t* arow = A + (size_t)(m0 + (lane & 15)) * K;
  // cooperative W staging: 256 lanes x 16B == 64 rows x 32 halves
  const int wrow = tid >> 2;
  const int wseg = (tid & 3) * 8;
  const half_t* wg = W + (size_t)(n0 + wrow) * K + wseg;
  half_t* wl = &wtile[wrow * 32 + wseg];

  v8f acc[4];
#pragma unroll
  for (int t = 0; t < 4; ++t) acc[t] = v8f{0, 0, 0, 0, 0, 0, 0, 0};

  for (int k0 = 0; k0 < K; k0 += 32) {
    ASYNC_CP16(wl, wg + k0);
    if (k0 + 32 < K) {
      __builtin_prefetch(arow + k0 + 32, 0, 1);
      __builtin_prefetch(wg + k0 + 32, 0, 1);
    }
    ASYNC_WAIT();
    __syncthreads();
    v16h af = ld_frag_row(arow, k0, lane);
#pragma unroll
    for (int t = 0; t < 4; ++t) {
      v16h bf = ld_frag_row(&wtile[(16 * t + (lane & 15)) * 32], 0, lane);
      acc[t] = __builtin_amdgcn_wmma_f32_16x16x32_f16(
          false, af, false, bf, (short)0, acc[t], false, false);
    }
    __syncthreads();
  }

  const int rbase = m0 + ((lane & 16) ? 8 : 0);
#pragma unroll
  for (int t = 0; t < 4; ++t) {
    const int col = n0 + 16 * t + (lane & 15);
    const float bv = bias ? bias[col] : 0.0f;
#pragma unroll
    for (int v = 0; v < 8; ++v) {
      const size_t idx = (size_t)(rbase + v) * Nout + col;
      float x = acc[t][v] + bv;
      if (GELU) x = 0.5f * x * (1.0f + erff(x * 0.70710678118654752f));
      if (RESID) x += resid[idx];
      if (OUT_F16) outH[idx] = (half_t)x;
      else         outF[idx] = x;
    }
  }
}

// ---- Flash attention ------------------------------------------------------
// One wave owns 16 query rows; the 8 waves of a block share one (b,h) and
// cooperatively stage K (32x64) and V^T (64x32) tiles in LDS per key block.
__global__ __launch_bounds__(256)
void attn_wmma_kernel(const half_t* __restrict__ qkvQ,
                      const half_t* __restrict__ qkvKV,
                      const half_t* __restrict__ vt,
                      half_t* __restrict__ out) {
  __shared__ __align__(16) half_t ktile[32 * 64];
  __shared__ __align__(16) half_t vtile[64 * 32];
  __shared__ __align__(16) half_t pbuf[8][16 * 32];
  const int tid = threadIdx.x;
  const int lane = tid & 31;
  const int wave = tid >> 5;
  const int wid = blockIdx.x * 8 + wave;
  const int qt = wid & 63;              // 64 query tiles per (b,h)
  const int bh = wid >> 6;              // block-uniform (64 % 8 == 0)
  const int b = bh / H_;
  const int h = bh % H_;
  const int ld = 3 * C_;

  const half_t* qbase =
      qkvQ + (size_t)(b * N_ + qt * 16 + (lane & 15)) * ld + h * D_;
  const v16h qf0 = ld_frag_row(qbase, 0, lane);
  const v16h qf1 = ld_frag_row(qbase, 32, lane);

  // staging source pointers (per-thread slice, 16B each)
  const int krow = tid >> 3, kseg = (tid & 7) * 8;     // K: 32 rows x 64
  const half_t* kg = qkvKV + (size_t)(b * N_ + krow) * ld + C_ + h * D_ + kseg;
  half_t* kl = &ktile[krow * 64 + kseg];
  const int vrow = tid >> 2, vseg = (tid & 3) * 8;     // V^T: 64 rows x 32
  const half_t* vg = vt + (size_t)(bh * D_ + vrow) * N_ + vseg;
  half_t* vl = &vtile[vrow * 32 + vseg];

  float Mst[8], Lst[8];
  v8f O[4];
#pragma unroll
  for (int v = 0; v < 8; ++v) { Mst[v] = -INFINITY; Lst[v] = 0.0f; }
#pragma unroll
  for (int t = 0; t < 4; ++t) O[t] = v8f{0, 0, 0, 0, 0, 0, 0, 0};

  half_t* pl = &pbuf[wave][0];

  for (int kb = 0; kb < N_; kb += 32) {
    ASYNC_CP16(kl, kg + (size_t)kb * ld);
    ASYNC_CP16(vl, vg + kb);
    if (kb + 32 < N_) {
      __builtin_prefetch(kg + (size_t)(kb + 32) * ld, 0, 1);
      __builtin_prefetch(vg + kb + 32, 0, 1);
    }
    ASYNC_WAIT();
    __syncthreads();

    v8f S[2];
#pragma unroll
    for (int st = 0; st < 2; ++st) {
      const half_t* kbase = &ktile[(16 * st + (lane & 15)) * 64];
      v16h kf0 = ld_frag_row(kbase, 0, lane);
      v16h kf1 = ld_frag_row(kbase, 32, lane);
      v8f s = v8f{0, 0, 0, 0, 0, 0, 0, 0};
      s = __builtin_amdgcn_wmma_f32_16x16x32_f16(false, qf0, false, kf0, (short)0, s, false, false);
      s = __builtin_amdgcn_wmma_f32_16x16x32_f16(false, qf1, false, kf1, (short)0, s, false, false);
      S[st] = s;
    }
    // Online softmax. C-layout: lanes 0-15 hold rows 0-7 (vgpr v), lanes
    // 16-31 rows 8-15 -> shfl_xor masks 1..8 stay inside each 16-lane half.
#pragma unroll
    for (int v = 0; v < 8; ++v) {
      float s0 = S[0][v] * 0.125f;   // 1/sqrt(64)
      float s1 = S[1][v] * 0.125f;
      float rm = fmaxf(s0, s1);
#pragma unroll
      for (int m = 1; m < 16; m <<= 1) rm = fmaxf(rm, __shfl_xor(rm, m, 32));
      const float nm = fmaxf(Mst[v], rm);
      const float corr = __expf(Mst[v] - nm);
      const float p0 = __expf(s0 - nm);
      const float p1 = __expf(s1 - nm);
      float rs = p0 + p1;
#pragma unroll
      for (int m = 1; m < 16; m <<= 1) rs += __shfl_xor(rs, m, 32);
      Lst[v] = Lst[v] * corr + rs;
      Mst[v] = nm;
#pragma unroll
      for (int t = 0; t < 4; ++t) O[t][v] *= corr;
      const int prow = v + ((lane & 16) ? 8 : 0);
      pl[prow * 32 + (lane & 15)]      = (half_t)p0;
      pl[prow * 32 + 16 + (lane & 15)] = (half_t)p1;
    }
    // P (16x32) back as A-fragment; V^T fragments from LDS (K-contiguous).
    v16h pf = ld_frag_row(pl + (lane & 15) * 32, 0, lane);
#pragma unroll
    for (int t = 0; t < 4; ++t) {
      v16h vf = ld_frag_row(&vtile[(16 * t + (lane & 15)) * 32], 0, lane);
      O[t] = __builtin_amdgcn_wmma_f32_16x16x32_f16(false, pf, false, vf, (short)0, O[t], false, false);
    }
    __syncthreads();
  }

#pragma unroll
  for (int v = 0; v < 8; ++v) {
    const float inv = 1.0f / Lst[v];
    const int grow = b * N_ + qt * 16 + v + ((lane & 16) ? 8 : 0);
#pragma unroll
    for (int t = 0; t < 4; ++t) {
      const int col = h * D_ + 16 * t + (lane & 15);
      out[(size_t)grow * C_ + col] = (half_t)(O[t][v] * inv);
    }
  }
}

// ---------------------------------------------------------------------------
extern "C" void kernel_launch(void* const* d_in, const int* in_sizes, int n_in,
                              void* d_out, int out_size, void* d_ws, size_t ws_size,
                              hipStream_t stream) {
  (void)in_sizes; (void)n_in; (void)out_size; (void)ws_size;
  const float* x1     = (const float*)d_in[0];
  const float* x2     = (const float*)d_in[1];
  const float* w_qkv  = (const float*)d_in[2];
  const float* w_proj = (const float*)d_in[3];
  const float* b_proj = (const float*)d_in[4];
  const float* g1     = (const float*)d_in[5];
  const float* b1     = (const float*)d_in[6];
  const float* g2     = (const float*)d_in[7];
  const float* b2     = (const float*)d_in[8];
  const float* w_fc1  = (const float*)d_in[9];
  const float* b_fc1  = (const float*)d_in[10];
  const float* w_fc2  = (const float*)d_in[11];
  const float* b_fc2  = (const float*)d_in[12];
  float* out = (float*)d_out;

  constexpr size_t sWqkv  = (size_t)3 * C_ * C_ * 2;
  constexpr size_t sWproj = (size_t)C_ * C_ * 2;
  constexpr size_t sWfc1  = (size_t)HID3 * C_ * 2;
  constexpr size_t sWfc2  = (size_t)C_ * HID3 * 2;
  constexpr size_t sN     = (size_t)M_ * C_ * 2;
  constexpr size_t sQKV   = (size_t)M_ * 3 * C_ * 2;
  constexpr size_t BNC    = (size_t)M_ * C_;

  char* ws = (char*)d_ws;
  size_t off = 0;
  half_t* wqkv_h  = (half_t*)(ws + off); off += sWqkv;
  half_t* wproj_h = (half_t*)(ws + off); off += sWproj;
  half_t* wfc1_h  = (half_t*)(ws + off); off += sWfc1;
  half_t* wfc2_h  = (half_t*)(ws + off); off += sWfc2;
  half_t* nA_h    = (half_t*)(ws + off); off += sN;    // LN outputs (reused for LN2)
  half_t* nB_h    = (half_t*)(ws + off); off += sN;
  half_t* qkvA_h  = (half_t*)(ws + off); off += sQKV;  // reused as MLP hidden
  half_t* qkvB_h  = (half_t*)(ws + off); off += sQKV;
  half_t* oA_h    = (half_t*)(ws + off); off += sN;
  half_t* oB_h    = (half_t*)(ws + off); off += sN;
  half_t* vtA_h   = (half_t*)(ws + off); off += sN;    // V transposed [B*H, D, N]
  half_t* vtB_h   = (half_t*)(ws + off); off += sN;
  half_t* h_h     = qkvA_h;   // 8192x3072 f16 fits across qkvA+qkvB region

  // 1) weights -> f16
  auto cvt = [&](const float* src, half_t* dst, int n) {
    f32_to_f16_kernel<<<(n + 255) / 256, 256, 0, stream>>>(src, dst, n);
  };
  cvt(w_qkv,  wqkv_h,  3 * C_ * C_);
  cvt(w_proj, wproj_h, C_ * C_);
  cvt(w_fc1,  wfc1_h,  HID3 * C_);
  cvt(w_fc2,  wfc2_h,  C_ * HID3);

  // 2) LN1 -> f16
  layernorm_f16_kernel<<<M_, 256, 0, stream>>>(x1, g1, b1, nA_h);
  layernorm_f16_kernel<<<M_, 256, 0, stream>>>(x2, g1, b1, nB_h);

  // 3) QKV GEMMs: [8192,768] x [2304,768]^T -> f16
  {
    const int blocks = (M_ / 128) * (3 * C_ / 64);   // 2304
    gemm_wmma_kernel<true, false, false><<<blocks, 256, 0, stream>>>(
        nA_h, wqkv_h, nullptr, nullptr, nullptr, qkvA_h, M_, 3 * C_, C_);
    gemm_wmma_kernel<true, false, false><<<blocks, 256, 0, stream>>>(
        nB_h, wqkv_h, nullptr, nullptr, nullptr, qkvB_h, M_, 3 * C_, C_);
  }

  // 4) V pre-transpose, then attention: self (q,k,v) and cross (q,k2,v2);
  //    q comes from branch 1 in both cases (per reference).
  {
    const int tblocks = (B_ * H_ * D_ * N_) / 256;   // 24576
    v_transpose_kernel<<<tblocks, 256, 0, stream>>>(qkvA_h, vtA_h);
    v_transpose_kernel<<<tblocks, 256, 0, stream>>>(qkvB_h, vtB_h);
    const int blocks = B_ * H_ * (N_ / 16) / 8;      // 768
    attn_wmma_kernel<<<blocks, 256, 0, stream>>>(qkvA_h, qkvA_h, vtA_h, oA_h);
    attn_wmma_kernel<<<blocks, 256, 0, stream>>>(qkvA_h, qkvB_h, vtB_h, oB_h);
  }

  // 5) proj + bias + residual -> f32 into d_out
  {
    const int blocks = (M_ / 128) * (C_ / 64);       // 768
    gemm_wmma_kernel<false, false, true><<<blocks, 256, 0, stream>>>(
        oA_h, wproj_h, b_proj, x1, out, nullptr, M_, C_, C_);
    gemm_wmma_kernel<false, false, true><<<blocks, 256, 0, stream>>>(
        oB_h, wproj_h, b_proj, x2, out + BNC, nullptr, M_, C_, C_);
  }

  // 6) MLP per branch: LN2 -> fc1(+gelu, f16) -> fc2(+bias+residual, in place)
  for (int br = 0; br < 2; ++br) {
    float* xr = out + (size_t)br * BNC;
    layernorm_f16_kernel<<<M_, 256, 0, stream>>>(xr, g2, b2, nA_h);
    {
      const int blocks = (M_ / 128) * (HID3 / 64);   // 3072
      gemm_wmma_kernel<true, true, false><<<blocks, 256, 0, stream>>>(
          nA_h, wfc1_h, b_fc1, nullptr, nullptr, h_h, M_, HID3, C_);
    }
    {
      const int blocks = (M_ / 128) * (C_ / 64);     // 768
      gemm_wmma_kernel<false, false, true><<<blocks, 256, 0, stream>>>(
          h_h, wfc2_h, b_fc2, xr, xr, nullptr, M_, C_, HID3);
    }
  }
}
